// CausalSelfAttention_55190329753734
// MI455X (gfx1250) — compile-verified, matrix operand order checked by reading, not run on hardware
//
#include <hip/hip_runtime.h>

#define B_ 2
#define T_ 2048
#define C_ 1024
#define NH 16
#define HD 64

typedef __attribute__((ext_vector_type(16))) __bf16 v16bf;
typedef __attribute__((ext_vector_type(8)))  float  v8f;

union FragU { uint4 u[2]; v16bf v; };

#if defined(__gfx1250__) && __has_builtin(__builtin_amdgcn_global_load_async_to_lds_b128)
#define HAVE_ASYNC_LDS 1
#else
#define HAVE_ASYNC_LDS 0
#endif

// Builtin signature (from hipcc diagnostic): param0 = v4i addrspace(1)*,
// param1 = v4i addrspace(3)*, then imm offset, imm cpol.
typedef int v4i_raw __attribute__((vector_size(16)));
typedef __attribute__((address_space(1))) v4i_raw* gv4_ptr;
typedef __attribute__((address_space(3))) v4i_raw* lv4_ptr;

// 16-byte global -> LDS copy: async (ASYNCcnt) on gfx1250, sync fallback else.
__device__ __forceinline__ void cp16(unsigned short* l, const unsigned short* g) {
#if HAVE_ASYNC_LDS
  __builtin_amdgcn_global_load_async_to_lds_b128(
      (gv4_ptr)(void*)const_cast<unsigned short*>(g), (lv4_ptr)(void*)l, 0, 0);
#else
  *(uint4*)l = *(const uint4*)g;
#endif
}

__device__ __forceinline__ void async_wait0() {
#if HAVE_ASYNC_LDS
#if __has_builtin(__builtin_amdgcn_s_wait_asynccnt)
  __builtin_amdgcn_s_wait_asynccnt(0);
#else
  asm volatile("s_wait_asynccnt 0" ::: "memory");
#endif
#endif
}

__device__ __forceinline__ unsigned short f2bf(float f) {
  union { float f; unsigned u; } x; x.f = f;
  unsigned r = (x.u + 0x7FFFu + ((x.u >> 16) & 1u)) >> 16;
  return (unsigned short)r;
}

// Load one 16x32 bf16 A/B fragment for lane (g = lane>>4).
// p points at a contiguous 32-element K-run for this lane's row/col.
__device__ __forceinline__ v16bf ld_frag(const unsigned short* p, int g) {
  FragU f;
  f.u[0] = *(const uint4*)(p + 8 * g);
  f.u[1] = *(const uint4*)(p + 16 + 8 * g);
  return f.v;
}

__device__ __forceinline__ v8f wmma_bf16(v16bf a, v16bf b, v8f c) {
  return __builtin_amdgcn_wmma_f32_16x16x32_bf16(false, a, false, b, (short)0, c,
                                                 false, false);
}

// ---------------------------------------------------------------------------
__global__ void cvt_bf16(const float* __restrict__ src,
                         unsigned short* __restrict__ dst, int n) {
  int i = blockIdx.x * 256 + threadIdx.x;
  if (i < n) dst[i] = f2bf(src[i]);
}

// src[R][Cc] fp32 -> dst[Cc][R] bf16
__global__ void cvt_transpose(const float* __restrict__ src,
                              unsigned short* __restrict__ dst, int R, int Cc) {
  int c = blockIdx.x * 16 + threadIdx.x;
  int r = blockIdx.y * 16 + threadIdx.y;
  if (r < R && c < Cc) dst[(size_t)c * R + r] = f2bf(src[(size_t)r * Cc + c]);
}

// ---------------------------------------------------------------------------
// Tiled bf16 WMMA GEMM:  out = A[M][K] * Bt[N][K]^T + bias
// Block tile 128x256, 8 waves as 2x4, each wave 64x64 (16 accum frags).
// Double-buffered LDS fed by async global->LDS copies.
// mode 0: fp32 out[M][N].  mode 1: bf16 scatter into Q/K/V [B,H,T,D].
__global__ __launch_bounds__(256) void gemm_bf16_wmma(
    const unsigned short* __restrict__ A, const unsigned short* __restrict__ Bt,
    const float* __restrict__ bias, float* __restrict__ outF,
    unsigned short* __restrict__ outQ, unsigned short* __restrict__ outK,
    unsigned short* __restrict__ outV, int M, int N, int K, int mode) {
  __shared__ __align__(16) unsigned short lA[2][128 * 32];  // 2 x 8 KB
  __shared__ __align__(16) unsigned short lB[2][256 * 32];  // 2 x 16 KB

  const int tid = threadIdx.x;
  const int lane = tid & 31;
  const int wv = tid >> 5;
  const int wm = wv >> 2;   // 0..1 : 64-row slab
  const int wn = wv & 3;    // 0..3 : 64-col slab
  const int m0 = blockIdx.x * 128;
  const int n0 = blockIdx.y * 256;
  const int g = lane >> 4;
  const int hl = lane & 15;

  v8f acc[4][4] = {};

  auto load_tile = [&](int buf, int k0) {
#pragma unroll
    for (int it = 0; it < 2; ++it) {  // A: 128x32 = 512 x uint4
      int idx = tid + it * 256;
      int r = idx >> 2, c = (idx & 3) * 8;
      cp16(&lA[buf][r * 32 + c], &A[(size_t)(m0 + r) * K + k0 + c]);
    }
#pragma unroll
    for (int it = 0; it < 4; ++it) {  // B: 256x32 = 1024 x uint4
      int idx = tid + it * 256;
      int r = idx >> 2, c = (idx & 3) * 8;
      cp16(&lB[buf][r * 32 + c], &Bt[(size_t)(n0 + r) * K + k0 + c]);
    }
  };

  load_tile(0, 0);

  for (int k0 = 0; k0 < K; k0 += 32) {
    const int cur = (k0 >> 5) & 1;
    async_wait0();
    __syncthreads();
    if (k0 + 32 < K) load_tile(cur ^ 1, k0 + 32);  // overlap with WMMAs

    v16bf af[4], bfr[4];
#pragma unroll
    for (int fm = 0; fm < 4; ++fm)
      af[fm] = ld_frag(&lA[cur][(wm * 64 + fm * 16 + hl) * 32], g);
#pragma unroll
    for (int fn = 0; fn < 4; ++fn)
      bfr[fn] = ld_frag(&lB[cur][(wn * 64 + fn * 16 + hl) * 32], g);
#pragma unroll
    for (int fm = 0; fm < 4; ++fm)
#pragma unroll
      for (int fn = 0; fn < 4; ++fn)
        acc[fm][fn] = wmma_bf16(af[fm], bfr[fn], acc[fm][fn]);
  }

  // Epilogue. C layout: lane -> n = lane&15, rows m = i + 8*(lane>>4).
#pragma unroll
  for (int fm = 0; fm < 4; ++fm) {
#pragma unroll
    for (int fn = 0; fn < 4; ++fn) {
      int ng = n0 + wn * 64 + fn * 16 + hl;
      float bv = bias[ng];
#pragma unroll
      for (int i = 0; i < 8; ++i) {
        int mg = m0 + wm * 64 + fm * 16 + 8 * g + i;
        float val = acc[fm][fn][i] + bv;
        if (mode == 0) {
          outF[(size_t)mg * N + ng] = val;
        } else {
          int sel = ng >> 10;  // q / k / v
          int cc = ng & 1023;
          int h = cc >> 6, d = cc & 63;
          int b = mg >> 11, t = mg & 2047;  // T_ = 2048
          unsigned short* dst = (sel == 0) ? outQ : ((sel == 1) ? outK : outV);
          dst[(((size_t)(b * NH + h)) * T_ + t) * HD + d] = f2bf(val);
        }
      }
    }
  }
}

// ---------------------------------------------------------------------------
// Flash attention: one block per (b*h, 128 q rows); 8 waves x 16 q rows.
__global__ __launch_bounds__(256) void attn_kernel(
    const unsigned short* __restrict__ Qp, const unsigned short* __restrict__ Kp,
    const unsigned short* __restrict__ Vp, unsigned short* __restrict__ Y) {
  __shared__ __align__(16) unsigned short lQ[128 * 64];    // [q][d]  16 KB
  __shared__ __align__(16) unsigned short lK[64 * 64];     // [t][d]   8 KB
  __shared__ __align__(16) unsigned short lVt[64 * 64];    // [d][t]   8 KB
  __shared__ __align__(16) unsigned short lP[8 * 16 * 64]; // per-wave 16 KB

  const int tid = threadIdx.x;
  const int lane = tid & 31;
  const int w = tid >> 5;  // wave 0..7
  const int g = lane >> 4;
  const int hl = lane & 15;
  const int bh = blockIdx.y;  // b*16 + h
  const int qt0 = blockIdx.x * 128;
  const size_t headBase = (size_t)bh * T_ * HD;

  // Stage Q tile (128x64 bf16) via async copies
#pragma unroll
  for (int it = 0; it < 4; ++it) {
    int idx = tid + it * 256;  // 0..1023
    int r = idx >> 3;
    int c0 = (idx & 7) * 8;
    cp16(&lQ[r * 64 + c0], &Qp[headBase + (size_t)(qt0 + r) * HD + c0]);
  }
  async_wait0();
  __syncthreads();

  v16bf qa0 = ld_frag(&lQ[(w * 16 + hl) * 64], g);       // d = 0..31
  v16bf qa1 = ld_frag(&lQ[(w * 16 + hl) * 64 + 32], g);  // d = 32..63

  float rmax[8], rsum[8];
#pragma unroll
  for (int i = 0; i < 8; ++i) { rmax[i] = -3.0e38f; rsum[i] = 0.f; }
  v8f o[4] = {};

  const int nkb = (qt0 >> 6) + 2;  // causal: k blocks covering [0, qt0+127]
  for (int kb = 0; kb < nkb; ++kb) {
    const int kt0 = kb * 64;
    __syncthreads();
    // K tile natural [t][d] (async); V tile transposed into [d][t] (manual)
#pragma unroll
    for (int it = 0; it < 2; ++it) {
      int idx = tid + it * 256;  // 0..511
      int r = idx >> 3;
      int c0 = (idx & 7) * 8;
      cp16(&lK[r * 64 + c0], &Kp[headBase + (size_t)(kt0 + r) * HD + c0]);
      uint4 pk = *(const uint4*)&Vp[headBase + (size_t)(kt0 + r) * HD + c0];
      const unsigned short* ps = (const unsigned short*)&pk;
#pragma unroll
      for (int j = 0; j < 8; ++j) lVt[(c0 + j) * 64 + r] = ps[j];
    }
    async_wait0();
    __syncthreads();

    // S = Q * K^T  (16 q rows x 64 k cols per wave)
    v8f s[4];
#pragma unroll
    for (int fn = 0; fn < 4; ++fn) {
      v8f z = {};
      v16bf kb0 = ld_frag(&lK[(fn * 16 + hl) * 64], g);
      v16bf kb1 = ld_frag(&lK[(fn * 16 + hl) * 64 + 32], g);
      z = wmma_bf16(qa0, kb0, z);
      z = wmma_bf16(qa1, kb1, z);
      s[fn] = z;
    }

    // mask + scale + row max (reduce over the 16 lanes of each half)
    float mrow[8];
#pragma unroll
    for (int i = 0; i < 8; ++i) {
      int qg = qt0 + w * 16 + 8 * g + i;
      float mv = -3.0e38f;
#pragma unroll
      for (int fn = 0; fn < 4; ++fn) {
        int kg = kt0 + fn * 16 + hl;
        float v = s[fn][i] * 0.125f;  // 1/sqrt(64)
        if (kg > qg) v = -3.0e38f;
        s[fn][i] = v;
        mv = fmaxf(mv, v);
      }
      for (int off = 1; off < 16; off <<= 1)
        mv = fmaxf(mv, __shfl_xor(mv, off, 32));
      mrow[i] = mv;
    }

    // online softmax update
    float alpha[8];
#pragma unroll
    for (int i = 0; i < 8; ++i) {
      float mn = fmaxf(rmax[i], mrow[i]);
      alpha[i] = __expf(rmax[i] - mn);
      rmax[i] = mn;
      float sum = 0.f;
#pragma unroll
      for (int fn = 0; fn < 4; ++fn) {
        float p = __expf(s[fn][i] - mn);
        s[fn][i] = p;
        sum += p;
      }
      for (int off = 1; off < 16; off <<= 1) sum += __shfl_xor(sum, off, 32);
      rsum[i] = rsum[i] * alpha[i] + sum;
    }

    // C-frag -> A-frag relayout of P through wave-private LDS
    unsigned short* pw = &lP[w * 16 * 64];
#pragma unroll
    for (int i = 0; i < 8; ++i) {
      int m = 8 * g + i;
#pragma unroll
      for (int fn = 0; fn < 4; ++fn)
        pw[m * 64 + fn * 16 + hl] = f2bf(s[fn][i]);
    }

    // rescale running output
#pragma unroll
    for (int fd = 0; fd < 4; ++fd)
#pragma unroll
      for (int i = 0; i < 8; ++i) o[fd][i] *= alpha[i];

    v16bf pa0 = ld_frag(&pw[hl * 64], g);       // k = 0..31
    v16bf pa1 = ld_frag(&pw[hl * 64 + 32], g);  // k = 32..63
#pragma unroll
    for (int fd = 0; fd < 4; ++fd) {
      v16bf vb0 = ld_frag(&lVt[(fd * 16 + hl) * 64], g);
      v16bf vb1 = ld_frag(&lVt[(fd * 16 + hl) * 64 + 32], g);
      o[fd] = wmma_bf16(pa0, vb0, o[fd]);
      o[fd] = wmma_bf16(pa1, vb1, o[fd]);
    }
  }

  // normalize + write y in [B,T,C] layout (heads re-interleaved), bf16
  const int b = bh >> 4, h = bh & 15;
#pragma unroll
  for (int fd = 0; fd < 4; ++fd) {
    int d = fd * 16 + hl;
#pragma unroll
    for (int i = 0; i < 8; ++i) {
      int qg = qt0 + w * 16 + 8 * g + i;
      float val = o[fd][i] / rsum[i];
      Y[((size_t)(b * T_ + qg)) * C_ + h * HD + d] = f2bf(val);
    }
  }
}

// ---------------------------------------------------------------------------
extern "C" void kernel_launch(void* const* d_in, const int* in_sizes, int n_in,
                              void* d_out, int out_size, void* d_ws,
                              size_t ws_size, hipStream_t stream) {
  (void)in_sizes; (void)n_in; (void)out_size; (void)ws_size;
  const float* x = (const float*)d_in[0];
  const float* w_attn = (const float*)d_in[1];
  const float* b_attn = (const float*)d_in[2];
  const float* w_proj = (const float*)d_in[3];
  const float* b_proj = (const float*)d_in[4];
  float* out = (float*)d_out;

  unsigned short* ws = (unsigned short*)d_ws;
  size_t off = 0;
  const size_t NTOK = (size_t)B_ * T_ * C_;  // 4,194,304
  unsigned short* xb = ws + off;  off += NTOK;
  unsigned short* waT = ws + off; off += (size_t)3 * C_ * C_;
  unsigned short* wpT = ws + off; off += (size_t)C_ * C_;
  unsigned short* q = ws + off;   off += NTOK;
  unsigned short* k = ws + off;   off += NTOK;
  unsigned short* v = ws + off;   off += NTOK;
  unsigned short* yb = ws + off;  off += NTOK;

  int nx = (int)NTOK;
  cvt_bf16<<<(nx + 255) / 256, 256, 0, stream>>>(x, xb, nx);
  cvt_transpose<<<dim3((3 * C_) / 16, C_ / 16), dim3(16, 16), 0, stream>>>(
      w_attn, waT, C_, 3 * C_);
  cvt_transpose<<<dim3(C_ / 16, C_ / 16), dim3(16, 16), 0, stream>>>(
      w_proj, wpT, C_, C_);

  // qkv = x @ w_attn + b_attn, scattered into per-head Q/K/V (bf16)
  gemm_bf16_wmma<<<dim3((B_ * T_) / 128, (3 * C_) / 256), 256, 0, stream>>>(
      xb, waT, b_attn, nullptr, q, k, v, B_ * T_, 3 * C_, C_, 1);

  // causal flash attention
  attn_kernel<<<dim3(T_ / 128, B_ * NH), 256, 0, stream>>>(q, k, v, yb);

  // out = y @ w_proj + b_proj (fp32)
  gemm_bf16_wmma<<<dim3((B_ * T_) / 128, C_ / 256), 256, 0, stream>>>(
      yb, wpT, b_proj, out, nullptr, nullptr, nullptr, B_ * T_, C_, C_, 0);
}
